// GPRGNN_88802743812567
// MI455X (gfx1250) — matrix operand order repaired and checked by reading, not compiled, not run on hardware
//
#include <hip/hip_runtime.h>
#include <hip/hip_bf16.h>

#define NNODE   50000
#define IN_DIM  512
#define HID_DIM 256
#define OUT_DIM 40
#define OUT_PAD 48
#define KHOPS   10
#define MT1     5      // row tiles per block, GEMM1 (80 rows, 625 blocks)
#define MT2     5      // row tiles per block, GEMM2

typedef __attribute__((ext_vector_type(16))) __bf16 v16bf;
typedef __attribute__((ext_vector_type(8)))  __bf16 v8bf;
typedef __attribute__((ext_vector_type(4)))  __bf16 v4bf;
typedef __attribute__((ext_vector_type(8)))  float  v8f;
typedef __attribute__((ext_vector_type(4)))  float  v4f;

__device__ __forceinline__ __bf16 f2bf(float f) {
    unsigned u = __builtin_bit_cast(unsigned, f);
    u = (u + 0x7FFFu + ((u >> 16) & 1u)) >> 16;
    return __builtin_bit_cast(__bf16, (unsigned short)u);
}

// ---------------- precision pre-pass: f32 -> bf16 ----------------
__global__ void k_cvt4(const float* __restrict__ src, __bf16* __restrict__ dst, long n) {
    long i = (blockIdx.x * (long)blockDim.x + threadIdx.x) * 4;
    if (i < n) {
        v4f s = *(const v4f*)(src + i);
        v4bf d;
        d[0] = f2bf(s[0]); d[1] = f2bf(s[1]); d[2] = f2bf(s[2]); d[3] = f2bf(s[3]);
        *(v4bf*)(dst + i) = d;
    }
}

// W2 [40,256] -> bf16 padded to [48,256] (zero rows 40..47) so GEMM2 B loads are unmasked
__global__ void k_cvt_w2(const float* __restrict__ W2, __bf16* __restrict__ W2p) {
    int i = blockIdx.x * blockDim.x + threadIdx.x;
    if (i < OUT_PAD * HID_DIM) {
        int r = i / HID_DIM;
        W2p[i] = (r < OUT_DIM) ? f2bf(W2[i])
                               : __builtin_bit_cast(__bf16, (unsigned short)0);
    }
}

// ---------------- edge normalization ----------------
__global__ void k_zero(float* __restrict__ p, int n) {
    int i = blockIdx.x * blockDim.x + threadIdx.x;
    if (i < n) p[i] = 0.f;
}

__global__ void k_deg(const int* __restrict__ row, float* __restrict__ deg, int e) {
    int i = blockIdx.x * blockDim.x + threadIdx.x;
    if (i < e) atomicAdd(&deg[row[i]], 1.0f);
}

__global__ void k_dinv(const float* __restrict__ deg, float* __restrict__ dinv, int n) {
    int i = blockIdx.x * blockDim.x + threadIdx.x;
    if (i < n) {
        float d = deg[i];
        d = d < 1.f ? 1.f : d;
        dinv[i] = rsqrtf(d);
    }
}

__global__ void k_norm(const int* __restrict__ row, const int* __restrict__ col,
                       const float* __restrict__ dinv, float* __restrict__ nrm, int e) {
    int i = blockIdx.x * blockDim.x + threadIdx.x;
    if (i < e) nrm[i] = dinv[row[i]] * dinv[col[i]];
}

// ---------------- GEMM1 + BN + ReLU (bf16 WMMA) ----------------
// 256 threads = 8 waves; block computes 80 rows x 256 cols.
// Each wave: 2 column tiles x 5 row tiles -> 10 WMMAs per K-step of 32.
__global__ __launch_bounds__(256) void mlp1_wmma(
    const __bf16* __restrict__ Xb, const __bf16* __restrict__ W1b,
    const float* __restrict__ b1, const float* __restrict__ bnw,
    const float* __restrict__ bnb, const float* __restrict__ bnm,
    const float* __restrict__ bnv, __bf16* __restrict__ H)
{
    const int wave = threadIdx.x >> 5;     // 0..7 -> col tiles wave*2, wave*2+1
    const int lane = threadIdx.x & 31;
    const int lo = lane & 15;              // A: row M; B/C/D: col N
    const int hi = lane >> 4;              // half-wave selector
    const int m0 = blockIdx.x * (16 * MT1);

    v8f acc[MT1][2] = {};

    for (int kb = 0; kb < IN_DIM; kb += 32) {
        // A fragments: per lane two contiguous 8-elem bf16 chunks
        // K = hi*8 + [0,8)  and  K = hi*8 + 16 + [0,8)
        v16bf a[MT1];
        #pragma unroll
        for (int rt = 0; rt < MT1; ++rt) {
            const __bf16* ap = Xb + (size_t)(m0 + rt * 16 + lo) * IN_DIM + kb + hi * 8;
            v8bf a0 = *(const v8bf*)ap;
            v8bf a1 = *(const v8bf*)(ap + 16);
            a[rt] = __builtin_shufflevector(a0, a1,
                     0,1,2,3,4,5,6,7,8,9,10,11,12,13,14,15);
        }
        #pragma unroll
        for (int t = 0; t < 2; ++t) {
            // B fragment: per lane two contiguous 8-elem bf16 chunks, K = hi*8 + [0,8) and hi*8 + 16 + [0,8)
            int hcol = (wave * 2 + t) * 16 + lo;
            const __bf16* bp = W1b + (size_t)hcol * IN_DIM + kb + hi * 8;
            v16bf b = __builtin_shufflevector(*(const v8bf*)bp, *(const v8bf*)(bp + 16),
                     0,1,2,3,4,5,6,7,8,9,10,11,12,13,14,15);
            #pragma unroll
            for (int rt = 0; rt < MT1; ++rt)
                acc[rt][t] = __builtin_amdgcn_wmma_f32_16x16x32_bf16(
                    false, a[rt], false, b, (short)0, acc[rt][t], false, false);
        }
    }

    // Epilogue: bias + BN(eval) + ReLU -> bf16 H [N, 256]
    #pragma unroll
    for (int t = 0; t < 2; ++t) {
        int hcol = (wave * 2 + t) * 16 + lo;            // C/D: n = lane&15
        float s    = bnw[hcol] * rsqrtf(bnv[hcol] + 1e-5f);
        float base = (b1[hcol] - bnm[hcol]) * s + bnb[hcol];
        #pragma unroll
        for (int rt = 0; rt < MT1; ++rt) {
            #pragma unroll
            for (int r = 0; r < 8; ++r) {
                int m = m0 + rt * 16 + r + hi * 8;      // C/D: m = vgpr + 8*hi
                float v = acc[rt][t][r] * s + base;
                v = v > 0.f ? v : 0.f;
                H[(size_t)m * HID_DIM + hcol] = f2bf(v);
            }
        }
    }
}

// ---------------- GEMM2 (bf16 WMMA), fused acc init ----------------
// 96 threads = 3 waves (one col tile each, OUT padded to 48); 5 row tiles/block.
__global__ __launch_bounds__(96) void mlp2_wmma(
    const __bf16* __restrict__ H, const __bf16* __restrict__ W2p,
    const float* __restrict__ b2, const float* __restrict__ gamma,
    float* __restrict__ Z, float* __restrict__ Acc)
{
    const int wave = threadIdx.x >> 5;     // 0..2 = column tile
    const int lane = threadIdx.x & 31;
    const int lo = lane & 15;
    const int hi = lane >> 4;
    const int m0 = blockIdx.x * (16 * MT2);

    const int ocol = wave * 16 + lo;       // < 48, padded weights -> no masking in loop

    v8f acc[MT2] = {};
    for (int kb = 0; kb < HID_DIM; kb += 32) {
        v16bf a[MT2];
        #pragma unroll
        for (int rt = 0; rt < MT2; ++rt) {
            const __bf16* ap = H + (size_t)(m0 + rt * 16 + lo) * HID_DIM + kb + hi * 8;
            v8bf a0 = *(const v8bf*)ap;
            v8bf a1 = *(const v8bf*)(ap + 16);
            a[rt] = __builtin_shufflevector(a0, a1,
                     0,1,2,3,4,5,6,7,8,9,10,11,12,13,14,15);
        }
        const __bf16* bp = W2p + (size_t)ocol * HID_DIM + kb + hi * 8;
        v16bf b = __builtin_shufflevector(*(const v8bf*)bp, *(const v8bf*)(bp + 16),
                 0,1,2,3,4,5,6,7,8,9,10,11,12,13,14,15);
        #pragma unroll
        for (int rt = 0; rt < MT2; ++rt)
            acc[rt] = __builtin_amdgcn_wmma_f32_16x16x32_bf16(
                false, a[rt], false, b, (short)0, acc[rt], false, false);
    }

    if (ocol < OUT_DIM) {
        float bias = b2[ocol];
        float g0   = gamma[0];
        #pragma unroll
        for (int rt = 0; rt < MT2; ++rt) {
            #pragma unroll
            for (int r = 0; r < 8; ++r) {
                int m = m0 + rt * 16 + r + hi * 8;
                float v = acc[rt][r] + bias;
                size_t idx = (size_t)m * OUT_DIM + ocol;
                Z[idx]   = v;
                Acc[idx] = g0 * v;   // fused acc = gamma[0] * z
            }
        }
    }
}

// ---------------- propagation ----------------
// 320 threads = 8 edges x 40 channels; edge data staged once through LDS.
__global__ __launch_bounds__(320) void k_scatter(
    const int* __restrict__ row, const int* __restrict__ col,
    const float* __restrict__ nrm, const float* __restrict__ hcur,
    float* __restrict__ hnext, int e)
{
    __shared__ int   s_r[8];
    __shared__ int   s_c[8];
    __shared__ float s_n[8];
    const int t  = threadIdx.x;
    const int e0 = blockIdx.x * 8;
    if (t < 8) {
        int ei = e0 + t;
        if (ei < e) { s_r[t] = row[ei]; s_c[t] = col[ei]; s_n[t] = nrm[ei]; }
    }
    __syncthreads();
    int le = t / OUT_DIM;          // 0..7
    int j  = t - le * OUT_DIM;     // 0..39
    if (e0 + le >= e) return;
    float v = s_n[le] * hcur[(size_t)s_r[le] * OUT_DIM + j];
    atomicAdd(&hnext[(size_t)s_c[le] * OUT_DIM + j], v);
}

__global__ void k_update(const float* __restrict__ hnext, const float* __restrict__ gamma,
                         int k, float* __restrict__ acc, float* __restrict__ hold, int n) {
    int i = blockIdx.x * blockDim.x + threadIdx.x;
    if (i < n) {
        acc[i] += gamma[k] * hnext[i];
        hold[i] = 0.f;   // old h becomes next step's zeroed destination
    }
}

// ---------------- launch ----------------
extern "C" void kernel_launch(void* const* d_in, const int* in_sizes, int n_in,
                              void* d_out, int out_size, void* d_ws, size_t ws_size,
                              hipStream_t stream) {
    const float* x     = (const float*)d_in[0];
    const int*   edge  = (const int*)d_in[1];
    const float* W1    = (const float*)d_in[2];
    const float* b1    = (const float*)d_in[3];
    const float* bnw   = (const float*)d_in[4];
    const float* bnb   = (const float*)d_in[5];
    const float* bnm   = (const float*)d_in[6];
    const float* bnv   = (const float*)d_in[7];
    const float* W2    = (const float*)d_in[8];
    const float* b2    = (const float*)d_in[9];
    const float* gamma = (const float*)d_in[10];

    const int N = NNODE;
    const int E = in_sizes[1] / 2;
    const int* row = edge;
    const int* col = edge + E;

    // workspace layout (all chunk sizes are multiples of 64 B)
    char* w = (char*)d_ws;
    __bf16* Xb  = (__bf16*)w;  w += (size_t)N * IN_DIM * 2;
    __bf16* W1b = (__bf16*)w;  w += (size_t)HID_DIM * IN_DIM * 2;
    __bf16* W2p = (__bf16*)w;  w += (size_t)OUT_PAD * HID_DIM * 2;
    __bf16* H   = (__bf16*)w;  w += (size_t)N * HID_DIM * 2;
    float*  deg  = (float*)w;  w += (size_t)N * 4;
    float*  dinv = (float*)w;  w += (size_t)N * 4;
    float*  nrm  = (float*)w;  w += (size_t)E * 4;
    float*  bufA = (float*)w;  w += (size_t)N * OUT_DIM * 4;
    float*  bufB = (float*)w;
    float*  acc  = (float*)d_out;

    // bf16 pre-pass
    const long nx  = (long)N * IN_DIM;
    const long nw1 = (long)HID_DIM * IN_DIM;
    k_cvt4<<<(unsigned)((nx / 4 + 255) / 256), 256, 0, stream>>>(x, Xb, nx);
    k_cvt4<<<(unsigned)((nw1 / 4 + 255) / 256), 256, 0, stream>>>(W1, W1b, nw1);
    k_cvt_w2<<<(OUT_PAD * HID_DIM + 255) / 256, 256, 0, stream>>>(W2, W2p);

    // edge normalization
    k_zero<<<(N + 255) / 256, 256, 0, stream>>>(deg, N);
    k_deg<<<(E + 255) / 256, 256, 0, stream>>>(row, deg, E);
    k_dinv<<<(N + 255) / 256, 256, 0, stream>>>(deg, dinv, N);
    k_norm<<<(E + 255) / 256, 256, 0, stream>>>(row, col, dinv, nrm, E);

    // MLP (WMMA bf16)
    mlp1_wmma<<<N / (16 * MT1), 256, 0, stream>>>(Xb, W1b, b1, bnw, bnb, bnm, bnv, H);
    mlp2_wmma<<<N / (16 * MT2), 96, 0, stream>>>(H, W2p, b2, gamma, bufA, acc);

    // GPR propagation: acc = sum_k gamma[k] * h_k
    const int NO = N * OUT_DIM;
    k_zero<<<(NO + 255) / 256, 256, 0, stream>>>(bufB, NO);

    float* cur = bufA;
    float* nxt = bufB;
    const unsigned sblocks = (unsigned)((E + 7) / 8);
    for (int k = 1; k <= KHOPS; ++k) {
        k_scatter<<<sblocks, 320, 0, stream>>>(row, col, nrm, cur, nxt, E);
        k_update<<<(NO + 255) / 256, 256, 0, stream>>>(nxt, gamma, k, acc, cur, NO);
        float* t = cur; cur = nxt; nxt = t;
    }
}
